// myLstm_33492154974557
// MI455X (gfx1250) — matrix operand order repaired and checked by reading, not compiled, not run on hardware
//
#include <hip/hip_runtime.h>
#include <cstdint>
#include <cstddef>

typedef __attribute__((ext_vector_type(16))) __bf16 v16bf;
typedef __attribute__((ext_vector_type(8)))  float  v8f;

#define B_   64
#define T_   512
#define I_   512
#define H_   1024
#define G4   4096      // 4*H
#define KTOT 1536      // I + H
#define CHK  64        // K-chunk per staging round
#define NK   24        // KTOT / CHK

// Dynamic LDS partition (bytes)
#define SW_OFF   0                       // weight slice: 64 cols x 1536 K bf16 = 196608
#define SA_OFF   196608                  // A staging:   2 bufs x 64 rows x 64 K bf16 = 16384
#define SG_OFF   (196608 + 16384)        // gates:       4 x 64 x 16 f32 = 16384
#define SMEM_SZ  (196608 + 16384 + 16384)   // 229376 B  (< 320 KB WGP LDS)

// ---------------------------------------------------------------------------
// Prep: build K-contiguous bf16 weight Wt[4H][KTOT] (= [U;V]^T) + bias[4H]
// ---------------------------------------------------------------------------
__global__ void lstm_prep_w(const float* __restrict__ Ui, const float* __restrict__ Uf,
                            const float* __restrict__ Uc, const float* __restrict__ Uo,
                            const float* __restrict__ Vi, const float* __restrict__ Vf,
                            const float* __restrict__ Vc, const float* __restrict__ Vo,
                            const float* __restrict__ bi, const float* __restrict__ bf,
                            const float* __restrict__ bc, const float* __restrict__ bo,
                            __bf16* __restrict__ Wt, float* __restrict__ bias)
{
    int gid = blockIdx.x * blockDim.x + threadIdx.x;      // covers G4*KTOT
    if (gid >= G4 * KTOT) return;
    int n = gid / KTOT;                                   // output column (gate-major)
    int k = gid - n * KTOT;                               // contraction index
    int g = n >> 10, j = n & 1023;
    const float* U[4] = {Ui, Uf, Uc, Uo};
    const float* V[4] = {Vi, Vf, Vc, Vo};
    float v = (k < I_) ? U[g][k * H_ + j] : V[g][(k - I_) * H_ + j];
    Wt[(size_t)n * KTOT + k] = (__bf16)v;
    if (gid < G4) {
        const float* bb[4] = {bi, bf, bc, bo};
        bias[gid] = bb[gid >> 10][gid & 1023];
    }
}

// ---------------------------------------------------------------------------
// Prep: x[B,T,I] f32 -> xT[T,B,I] bf16 ; zero h0 (bf16), c (f32), barrier ctr
// ---------------------------------------------------------------------------
__global__ void lstm_prep_x(const float* __restrict__ x, __bf16* __restrict__ xT,
                            __bf16* __restrict__ h0, float* __restrict__ c0,
                            unsigned* __restrict__ counter)
{
    int gid = blockIdx.x * blockDim.x + threadIdx.x;      // B*T*I
    if (gid >= B_ * T_ * I_) return;
    int k = gid & (I_ - 1);
    int t = (gid >> 9) & (T_ - 1);
    int m = gid >> 18;
    xT[((size_t)t * B_ + m) * I_ + k] = (__bf16)x[gid];
    if (gid < B_ * H_) { h0[gid] = (__bf16)0.0f; c0[gid] = 0.0f; }
    if (gid == 0) *counter = 0u;
}

// ---------------------------------------------------------------------------
// CDNA5 async memory->LDS copy (no VGPR round trip, tracked by ASYNCcnt)
// ---------------------------------------------------------------------------
__device__ __forceinline__ void async_copy_b128(uint32_t lds_addr, const void* gaddr)
{
    asm volatile("global_load_async_to_lds_b128 %0, %1, off"
                 :: "v"(lds_addr), "v"(gaddr) : "memory");
}
__device__ __forceinline__ void wait_asynccnt0()
{
    asm volatile("s_wait_asynccnt 0x0" ::: "memory");
}

// ---------------------------------------------------------------------------
// Persistent recurrence kernel: 64 blocks (one per WGP), 512 time steps
// in-kernel. Block owns hidden slice j0..j0+15 for all 4 gates; its 192 KB
// weight slice lives in LDS for the whole kernel. A-chunks stream in via
// GLOBAL_LOAD_ASYNC_TO_LDS_B128 double buffering; device-wide step barrier
// via agent-scope atomic counter + s_sleep spin.
// ---------------------------------------------------------------------------
__global__ __launch_bounds__(256)
void lstm_persist(const __bf16* __restrict__ xT, const __bf16* __restrict__ Wt,
                  const float* __restrict__ bias, __bf16* __restrict__ hbf,
                  float* __restrict__ c_st, float* __restrict__ out_hs,
                  unsigned* __restrict__ counter)
{
    extern __shared__ __align__(16) unsigned char smem[];
    __bf16* sW = (__bf16*)(smem + SW_OFF);   // [64 cols][1536 K], K-contiguous
    __bf16* sA = (__bf16*)(smem + SA_OFF);   // [2][64 rows][64 K]
    float*  sG = (float*) (smem + SG_OFF);   // [4][64][16]

    const int tid  = threadIdx.x;
    const int lane = tid & 31;
    const int wave = tid >> 5;
    const int j0   = blockIdx.x * 16;

    const int mTile = wave & 3;
    const int g0    = (wave >> 2) * 2;
    const int g1    = g0 + 1;

    // ---- one-time: pull this block's weight slice into LDS (pure b128 copies)
    for (int c = wave; c < 64; c += 8) {
        int n = (c >> 4) * H_ + j0 + (c & 15);
        const uint4* src = (const uint4*)(Wt + (size_t)n * KTOT);
        uint4*       dst = (uint4*)(sW + c * KTOT);
        #pragma unroll
        for (int u = lane; u < KTOT / 8; u += 32) dst[u] = src[u];
    }
    const float bv0 = bias[g0 * H_ + j0 + (lane & 15)];
    const float bv1 = bias[g1 * H_ + j0 + (lane & 15)];
    __syncthreads();

    // per-thread staging coordinates (A chunk: 64 rows x 64 K, 2x b128/thread)
    const int stM  = tid >> 2;
    const int stKK = (tid & 3) * 16;
    const uint32_t stLds = (uint32_t)(uintptr_t)&sA[stM * CHK + stKK];
    const uint32_t bufStride = (uint32_t)(B_ * CHK * sizeof(__bf16));

    // Contiguous lane layouts (per 32-wide K window): half h holds K 16h..16h+15
    const int rowA  = mTile * 16 + (lane & 15);
    const int kbA   = (lane >= 16) ? 16 : 0;
    const int colB0 = (g0 * 16 + (lane & 15)) * KTOT + ((lane >= 16) ? 16 : 0);
    const int colB1 = (g1 * 16 + (lane & 15)) * KTOT + ((lane >= 16) ? 16 : 0);

    for (int t = 0; t < T_; ++t) {
        const __bf16* h_in  = hbf + (size_t)(t & 1) * (B_ * H_);
        __bf16*       h_out = hbf + (size_t)((t + 1) & 1) * (B_ * H_);

        v8f acc0 = {};
        v8f acc1 = {};

        auto stage = [&](int kc, int buf) {      // async: global -> LDS, no VGPRs
            const int k0 = kc * CHK;
            const __bf16* src = (k0 < I_)
                ? xT + ((size_t)t * B_ + stM) * I_ + (k0 + stKK)
                : h_in + stM * H_ + (k0 - I_) + stKK;
            const uint32_t dst = stLds + (uint32_t)buf * bufStride;
            async_copy_b128(dst,      src);
            async_copy_b128(dst + 16, src + 8);
        };

        stage(0, 0);
        wait_asynccnt0();
        __syncthreads();

        for (int kc = 0; kc < NK; ++kc) {
            const int buf = kc & 1;
            if (kc + 1 < NK) stage(kc + 1, buf ^ 1);

            #pragma unroll
            for (int s = 0; s < 2; ++s) {        // two 32-wide K sub-steps
                v16bf a, b0, b1;
                const __bf16* pa  = &sA[(buf * B_ + rowA) * CHK + s * 32];
                const __bf16* pb0 = &sW[colB0 + kc * CHK + s * 32];
                const __bf16* pb1 = &sW[colB1 + kc * CHK + s * 32];
                #pragma unroll
                for (int i = 0; i < 8; ++i) {    // A: K {kb..kb+15}
                    a[i]     = pa[kbA + i];
                    a[8 + i] = pa[kbA + 8 + i];
                }
                #pragma unroll
                for (int i = 0; i < 16; ++i) {   // B: 16 consecutive K per lane
                    b0[i] = pb0[i];
                    b1[i] = pb1[i];
                }
                acc0 = __builtin_amdgcn_wmma_f32_16x16x32_bf16(false, a, false, b0,
                                                               (short)0, acc0, false, false);
                acc1 = __builtin_amdgcn_wmma_f32_16x16x32_bf16(false, a, false, b1,
                                                               (short)0, acc1, false, false);
            }
            wait_asynccnt0();                    // drain async fill of buf^1
            __syncthreads();
        }

        // bias + scatter gate tiles (C/D layout: VGPR r -> M = r (+8 hi half))
        const int mBase = mTile * 16 + ((lane >= 16) ? 8 : 0);
        const int jj    = lane & 15;
        #pragma unroll
        for (int r = 0; r < 8; ++r) {
            sG[(size_t)g0 * (B_ * 16) + (mBase + r) * 16 + jj] = acc0[r] + bv0;
            sG[(size_t)g1 * (B_ * 16) + (mBase + r) * 16 + jj] = acc1[r] + bv1;
        }
        __syncthreads();

        // fused LSTM elementwise update for this block's (m, j0+q) slice
        #pragma unroll
        for (int it = 0; it < 4; ++it) {
            int idx = tid + it * 256;            // 0..1023 = 64 batch x 16 j
            int m = idx >> 4;
            int q = idx & 15;
            float gi = sG[0 * (B_ * 16) + m * 16 + q];
            float gf = sG[1 * (B_ * 16) + m * 16 + q];
            float gc = sG[2 * (B_ * 16) + m * 16 + q];
            float go = sG[3 * (B_ * 16) + m * 16 + q];
            gi = 1.0f / (1.0f + __expf(-gi));
            gf = 1.0f / (1.0f + __expf(-gf));
            go = 1.0f / (1.0f + __expf(-go));
            gc = tanhf(gc);
            int j = j0 + q;
            float cold = c_st[m * H_ + j];
            float cnew = gf * cold + gi * gc;
            float hnew = go * tanhf(cnew);
            c_st[m * H_ + j]  = cnew;
            h_out[m * H_ + j] = (__bf16)hnew;
            out_hs[((size_t)m * T_ + t) * H_ + j] = hnew;   // hidden_seq [B,T,H]
        }

        // warm L2/WGP$ for next step's x slab (global_prefetch_b8)
        if (t + 1 < T_)
            __builtin_prefetch(xT + (size_t)(t + 1) * B_ * I_ + tid * 128, 0, 3);

        // ---- device-wide step barrier (monotonic counter, agent scope) ----
        __syncthreads();
        if (tid == 0) {
            __threadfence();                     // release h_out / c stores
            __hip_atomic_fetch_add(counter, 1u, __ATOMIC_RELEASE,
                                   __HIP_MEMORY_SCOPE_AGENT);
            const unsigned target = 64u * (unsigned)(t + 1);
            while (__hip_atomic_load(counter, __ATOMIC_ACQUIRE,
                                     __HIP_MEMORY_SCOPE_AGENT) < target)
                __builtin_amdgcn_s_sleep(2);
        }
        __syncthreads();
    }
}

// ---------------------------------------------------------------------------
// Tail outputs: h_T (copy of last hidden slice) and c_T
// ---------------------------------------------------------------------------
__global__ void lstm_final(const float* __restrict__ out_hs, const float* __restrict__ c_st,
                           float* __restrict__ out_tail)
{
    int gid = blockIdx.x * blockDim.x + threadIdx.x;  // B*H
    if (gid >= B_ * H_) return;
    int m = gid >> 10, j = gid & 1023;
    out_tail[gid]           = out_hs[((size_t)m * T_ + (T_ - 1)) * H_ + j];
    out_tail[B_ * H_ + gid] = c_st[gid];
}

// ---------------------------------------------------------------------------
extern "C" void kernel_launch(void* const* d_in, const int* in_sizes, int n_in,
                              void* d_out, int out_size, void* d_ws, size_t ws_size,
                              hipStream_t stream)
{
    const float* x  = (const float*)d_in[0];
    const float* Ui = (const float*)d_in[1];
    const float* Vi = (const float*)d_in[2];
    const float* bi = (const float*)d_in[3];
    const float* Uf = (const float*)d_in[4];
    const float* Vf = (const float*)d_in[5];
    const float* bfp= (const float*)d_in[6];
    const float* Uc = (const float*)d_in[7];
    const float* Vc = (const float*)d_in[8];
    const float* bc = (const float*)d_in[9];
    const float* Uo = (const float*)d_in[10];
    const float* Vo = (const float*)d_in[11];
    const float* bo = (const float*)d_in[12];

    // Workspace layout (256B-aligned), total ~44.6 MiB
    uint8_t* ws = (uint8_t*)d_ws;
    __bf16*   Wt   = (__bf16*)(ws);                        // 4096*1536*2 = 12,582,912
    __bf16*   xT   = (__bf16*)(ws + 12582912);             // 512*64*512*2 = 33,554,432
    float*    bias = (float*) (ws + 46137344);             // 4096*4 = 16,384
    __bf16*   hbf  = (__bf16*)(ws + 46153728);             // 2*64*1024*2 = 262,144
    float*    cst  = (float*) (ws + 46415872);             // 64*1024*4 = 262,144
    unsigned* ctr  = (unsigned*)(ws + 46678016);           // 4

    float* out_hs   = (float*)d_out;
    float* out_tail = (float*)d_out + (size_t)B_ * T_ * H_;

    lstm_prep_w<<<(G4 * KTOT) / 256, 256, 0, stream>>>(
        Ui, Uf, Uc, Uo, Vi, Vf, Vc, Vo, bi, bfp, bc, bo, Wt, bias);

    lstm_prep_x<<<(B_ * T_ * I_) / 256, 256, 0, stream>>>(x, xT, hbf, cst, ctr);

    lstm_persist<<<64, 256, SMEM_SZ, stream>>>(xT, Wt, bias, hbf, cst, out_hs, ctr);

    lstm_final<<<(B_ * H_) / 256, 256, 0, stream>>>(out_hs, cst, out_tail);
}
